// SegmentedPoolingEncoderModel_32753420599620
// MI455X (gfx1250) — compile-verified, matrix operand order checked by reading, not run on hardware
//
#include <hip/hip_runtime.h>

// ---------------------------------------------------------------------------
// CDNA5 (gfx1250) fused SegmentedPoolingEncoder:
//   z = relu(x @ W1 + b1) @ W2 + b2 ;  out[s] = mean_{rows in seg s} z[row]
// Both GEMMs run on v_wmma_f32_16x16x32_bf16 (wave32). The x tile is DMA'd
// into LDS by the Tensor Data Mover (tensor_load_to_lds + s_wait_tensorcnt),
// h stays in LDS, ragged mean via ds_add_f32 + global_atomic_add_f32.
// ---------------------------------------------------------------------------

typedef __attribute__((ext_vector_type(16))) __bf16       v16bf;
typedef __attribute__((ext_vector_type(8)))  float        v8f;
typedef __attribute__((ext_vector_type(4)))  unsigned int u32x4;
typedef __attribute__((ext_vector_type(8)))  int          i32x8;
typedef __attribute__((ext_vector_type(4)))  int          i32x4;

#define T_TOK   16384
#define K1      512     // nelem
#define HID     2048    // hidden
#define N2      512     // latent
#define NSEG    16
#define TM      64      // rows per block
#define HC      64      // hidden chunk per iteration

#if __has_builtin(__builtin_amdgcn_tensor_load_to_lds) && \
    __has_builtin(__builtin_amdgcn_s_wait_tensorcnt)
#define HAVE_TDM 1
#endif

__device__ __forceinline__ unsigned short f2bf(float f) {
    union { float f; unsigned int u; } v; v.f = f;
    unsigned int u = v.u;
    unsigned int r = u + 0x7FFFu + ((u >> 16) & 1u);  // round-to-nearest-even
    return (unsigned short)(r >> 16);
}

// ---------------------------------------------------------------------------
// Pack f32 W[K][N] (row-major) into bf16 WMMA B-fragment layout:
// Wp[ntile][ktile][lane][16]; lane = (khalf<<4)|ncol holds 16 consecutive K
// values of column (ntile*16+ncol), K-half khalf (same contiguous k order as the A fragments).
// ---------------------------------------------------------------------------
__global__ void pack_w_kernel(const float* __restrict__ W,
                              unsigned short* __restrict__ Wp,
                              int K, int N) {
    const int nt = blockIdx.x;          // N/16 tiles
    const int kt = blockIdx.y;          // K/32 tiles
    const size_t KT = (size_t)(K >> 5);
    for (int e = threadIdx.x; e < 512; e += 256) {
        const int ln = e >> 4;
        const int j  = e & 15;
        const int n  = ln & 15;
        const int kh = ln >> 4;
        const int k  = kt * 32 + kh * 16 + j;
        const int c  = nt * 16 + n;
        const size_t dst = ((((size_t)nt * KT + kt) * 32) + ln) * 16 + j;
        Wp[dst] = f2bf(W[(size_t)k * N + c]);
    }
}

__global__ void zero_out_kernel(float* __restrict__ out, int n,
                                int* __restrict__ cnt) {
    const int i = blockIdx.x * blockDim.x + threadIdx.x;
    if (i < n)    out[i] = 0.0f;
    if (i < NSEG) cnt[i] = 0;
}

__global__ void count_kernel(const int* __restrict__ segid,
                             int* __restrict__ cnt) {
    const int i = blockIdx.x * blockDim.x + threadIdx.x;
    if (i < T_TOK) atomicAdd(&cnt[segid[i]], 1);
}

__global__ void finalize_kernel(float* __restrict__ out,
                                const int* __restrict__ cnt) {
    const int i = blockIdx.x * blockDim.x + threadIdx.x;   // 0..NSEG*N2-1
    const int s = i >> 9;
    out[i] = out[i] / fmaxf((float)cnt[s], 1.0f);
}

// ---------------------------------------------------------------------------
// Main fused kernel: 256 threads = 8 wave32s, 64 rows per block.
// Dynamic LDS:
//   xs[64][512] bf16   @      0 (65536 B) : x tile, A-fragment layout
//   hs[64][ 64] bf16   @  65536 ( 8192 B) : relu hidden-chunk staging
//   segacc[16][512]f32 @  73728 (32768 B) : per-block segment partial sums
//   sseg[64] int       @ 106496 (  256 B)
//   stag[16][512] f32  @ 106752 (32768 B) : TDM staging (raw f32 rows)
// ---------------------------------------------------------------------------
__launch_bounds__(256)
__global__ void fused_mlp_segpool(const float* __restrict__ x,
                                  const int* __restrict__ segid,
                                  const unsigned short* __restrict__ W1p,
                                  const float* __restrict__ b1g,
                                  const unsigned short* __restrict__ W2p,
                                  const float* __restrict__ b2g,
                                  float* __restrict__ out) {
    extern __shared__ char smem_raw[];
    unsigned short* xs     = (unsigned short*)smem_raw;
    unsigned short* hs     = (unsigned short*)(smem_raw + 65536);
    float*          segacc = (float*)(smem_raw + 73728);
    int*            sseg   = (int*)(smem_raw + 106496);
    float*          stag   = (float*)(smem_raw + 106752);

    const int tid   = threadIdx.x;
    const int lane  = tid & 31;
    const int wave  = tid >> 5;
    const int mrow  = lane & 15;   // M (or N) index within a 16-tile
    const int mhalf = lane >> 4;   // K-half selector
    const int row0  = blockIdx.x * TM;

    for (int i = tid; i < NSEG * N2; i += 256) segacc[i] = 0.0f;
    if (tid < TM) sseg[tid] = segid[row0 + tid];

    // ---- Prologue: bring the 64x512 f32 x tile into bf16 A-layout LDS -----
#ifdef HAVE_TDM
    {
        typedef __attribute__((address_space(3))) char lds_char;
        const unsigned stag_lds = (unsigned)(unsigned long long)(lds_char*)(void*)stag;
        for (int c = 0; c < 4; ++c) {                  // 4 chunks of 16 rows
            if (wave == 0) {
                const unsigned long long ga =
                    (unsigned long long)(const void*)(x + ((size_t)row0 + c * 16) * K1);
                u32x4 g0;
                g0[0] = 1u;                                          // count=1
                g0[1] = stag_lds;                                    // lds_addr
                g0[2] = (unsigned)ga;                                // addr[31:0]
                g0[3] = (unsigned)((ga >> 32) & 0x01FFFFFFu) | (2u << 30); // addr[56:32]|type=2
                i32x8 g1;
                g1[0] = (int)(2u << 16);        // data_size = 4 bytes
                g1[1] = (int)(512u << 16);      // tensor_dim0 = 512
                g1[2] = (int)(16u << 16);       // tensor_dim1 = 16
                g1[3] = (int)(512u << 16);      // tile_dim0 = 512
                g1[4] = 16;                     // tile_dim1 = 16
                g1[5] = 512;                    // tensor_dim0_stride = 512
                g1[6] = 0;
                g1[7] = 0;
                const i32x4 z4 = {0, 0, 0, 0};
#if defined(__clang_major__) && (__clang_major__ >= 23)
                const i32x8 z8 = {0, 0, 0, 0, 0, 0, 0, 0};
                __builtin_amdgcn_tensor_load_to_lds(g0, g1, z4, z4, z8, 0);
#else
                __builtin_amdgcn_tensor_load_to_lds(g0, g1, z4, z4, 0);
#endif
                __builtin_amdgcn_s_wait_tensorcnt(0);
            }
            __syncthreads();                           // data visible to all waves
            const float4* sg = (const float4*)stag;
            for (int i = tid; i < 16 * K1 / 4; i += 256) {
                float4 v = sg[i];
                const int o = c * 16 * K1 + i * 4;
                xs[o + 0] = f2bf(v.x); xs[o + 1] = f2bf(v.y);
                xs[o + 2] = f2bf(v.z); xs[o + 3] = f2bf(v.w);
            }
            __syncthreads();                           // staging reusable
        }
    }
#else
    {
        const float4* xg = (const float4*)(x + (size_t)row0 * K1);
        for (int i = tid; i < TM * K1 / 4; i += 256) {
            float4 v = xg[i];
            const int o = i * 4;
            xs[o + 0] = f2bf(v.x); xs[o + 1] = f2bf(v.y);
            xs[o + 2] = f2bf(v.z); xs[o + 3] = f2bf(v.w);
        }
        __syncthreads();
    }
#endif

    const v8f vz = {0.f, 0.f, 0.f, 0.f, 0.f, 0.f, 0.f, 0.f};
    v8f acc[4][4];                       // [nsubZ][msub] persistent z tiles
#pragma unroll
    for (int n = 0; n < 4; ++n)
#pragma unroll
        for (int m = 0; m < 4; ++m) acc[n][m] = vz;

    // ---- Hoisted base pointers: unrolled offsets become byte immediates ---
    const int msubA  = wave >> 1;              // phase-A M subtile of this wave
    const int nsubH0 = (wave & 1) * 2;         // first of two hidden col tiles
    // A fragments of x:  + k*32 halves
    const unsigned short* xabase = xs + (msubA * 16 + mrow) * K1 + mhalf * 16;
    // W1 fragments: idx = hc*32768 + (nsubH0+j)*8192 + k*512 + lane*16
    const unsigned short* w1w = W1p + (size_t)nsubH0 * 8192 + (size_t)lane * 16;
    // W2 fragments: idx = wave*131072 + n*32768 + hc*1024 + kk*512 + lane*16
    const unsigned short* w2w = W2p + (size_t)wave * 131072 + (size_t)lane * 16;
    // h A fragments: + m*1024 + kk*32 halves
    const unsigned short* habase = hs + mrow * HC + mhalf * 16;
    // h store base:  + i*64 + j*16 halves
    unsigned short* hsw = hs + (msubA * 16 + 8 * mhalf) * HC + nsubH0 * 16 + mrow;

    for (int hc = 0; hc < HID / HC; ++hc) {
        const unsigned short* w1c = w1w + (size_t)hc * 32768;
        const unsigned short* w2c = w2w + (size_t)hc * 1024;
        if (hc + 1 < HID / HC)
            __builtin_prefetch(w1c + 32768, 0, 1);     // global_prefetch next W1 chunk

        // ---- Phase A: hs = relu(x_tile @ W1[:, hc*64 .. +64] + b1) --------
        // One A fragment feeds both of this wave's column tiles.
        v8f c0 = vz, c1 = vz;
#pragma unroll
        for (int k = 0; k < 16; ++k) {                 // K1 = 512 = 16 * 32
            v16bf a  = *(const v16bf*)(xabase + k * 32);
            v16bf f0 = *(const v16bf*)(w1c + k * 512);
            v16bf f1 = *(const v16bf*)(w1c + 8192 + k * 512);
            c0 = __builtin_amdgcn_wmma_f32_16x16x32_bf16(false, a, false, f0,
                                                         (short)0, c0, false, false);
            c1 = __builtin_amdgcn_wmma_f32_16x16x32_bf16(false, a, false, f1,
                                                         (short)0, c1, false, false);
        }
        const float bb0 = b1g[hc * HC + nsubH0 * 16 + mrow];
        const float bb1 = b1g[hc * HC + nsubH0 * 16 + 16 + mrow];
#pragma unroll
        for (int i = 0; i < 8; ++i) {
            float v0 = c0[i] + bb0; v0 = v0 > 0.0f ? v0 : 0.0f;
            float v1 = c1[i] + bb1; v1 = v1 > 0.0f ? v1 : 0.0f;
            hsw[i * HC]      = f2bf(v0);
            hsw[i * HC + 16] = f2bf(v1);
        }
        __syncthreads();

        // ---- Phase B: acc += hs @ W2[hc*64.., wave's 64 columns] ----------
#pragma unroll
        for (int kk = 0; kk < 2; ++kk) {               // HC = 64 = 2 * 32
            v16bf a[4];
#pragma unroll
            for (int m = 0; m < 4; ++m)
                a[m] = *(const v16bf*)(habase + m * 16 * HC + kk * 32);
#pragma unroll
            for (int n = 0; n < 4; ++n) {
                v16bf f = *(const v16bf*)(w2c + (size_t)n * 32768 + kk * 512);
#pragma unroll
                for (int m = 0; m < 4; ++m)
                    acc[n][m] = __builtin_amdgcn_wmma_f32_16x16x32_bf16(
                            false, a[m], false, f, (short)0, acc[n][m], false, false);
            }
        }
        __syncthreads();
    }

    // ---- Epilogue: z = acc + b2, segment-sum in LDS, flush via atomics ----
#pragma unroll
    for (int n = 0; n < 4; ++n) {
        const int col = wave * 64 + n * 16 + mrow;
        const float bb = b2g[col];
#pragma unroll
        for (int m = 0; m < 4; ++m) {
#pragma unroll
            for (int i = 0; i < 8; ++i) {
                const int r = m * 16 + i + 8 * mhalf;
                atomicAdd(&segacc[sseg[r] * N2 + col], acc[n][m][i] + bb); // ds_add_f32
            }
        }
    }
    __syncthreads();
    for (int i = tid; i < NSEG * N2; i += 256) {
        const float v = segacc[i];
        if (v != 0.0f) atomicAdd(&out[i], v);          // global_atomic_add_f32
    }
}

// ---------------------------------------------------------------------------
extern "C" void kernel_launch(void* const* d_in, const int* in_sizes, int n_in,
                              void* d_out, int out_size, void* d_ws, size_t ws_size,
                              hipStream_t stream) {
    (void)in_sizes; (void)n_in; (void)out_size; (void)ws_size;
    const float* x   = (const float*)d_in[0];
    const int*   seg = (const int*)d_in[1];
    const float* W1  = (const float*)d_in[2];
    const float* b1  = (const float*)d_in[3];
    const float* W2  = (const float*)d_in[4];
    const float* b2  = (const float*)d_in[5];
    float* out = (float*)d_out;

    unsigned short* W1p = (unsigned short*)d_ws;                  // 2 MB
    unsigned short* W2p = W1p + (size_t)K1 * HID;                 // 2 MB
    int* cnt            = (int*)(W2p + (size_t)HID * N2);

    pack_w_kernel<<<dim3(HID / 16, K1 / 32), 256, 0, stream>>>(W1, W1p, K1, HID);
    pack_w_kernel<<<dim3(N2 / 16, HID / 32), 256, 0, stream>>>(W2, W2p, HID, N2);

    zero_out_kernel<<<(NSEG * N2 + 255) / 256, 256, 0, stream>>>(out, NSEG * N2, cnt);
    count_kernel<<<T_TOK / 256, 256, 0, stream>>>(seg, cnt);

    const size_t smem = 106752 + 32768;   // 136.25 KB dynamic LDS
    fused_mlp_segpool<<<T_TOK / TM, 256, smem, stream>>>(x, seg, W1p, b1, W2p, b2, out);

    finalize_kernel<<<(NSEG * N2) / 256, 256, 0, stream>>>(out, cnt);
}
